// SceneFlowPWC_17755394801920
// MI455X (gfx1250) — matrix operand order repaired and checked by reading, not compiled, not run on hardware
//
#include <hip/hip_runtime.h>
#include <hip/hip_bf16.h>

typedef _Float16 v8h  __attribute__((ext_vector_type(8)));
typedef _Float16 v16h __attribute__((ext_vector_type(16)));
typedef float    v8f  __attribute__((ext_vector_type(8)));

#define NPTS  8192
#define BATCH 2
#define DFEAT 64
#define HID   128
#define OUTC  64
#define KNN   16
#define KPAD  96   // 3 + 64 = 67 padded to 3*32
#define WVB   4    // waves per block in the MLP kernel

#define BIGF 3.0e38f

// ---------------------------------------------------------------------------
// Kernel A: xyz1_to_2 = xyz1 + flow1; 3-NN of each xyz2 point among xyz1_to_2;
// inverse-distance interpolate flow1 -> flow2; warped_xyz2 = xyz2 - flow2.
// Writes warped both to workspace ([B][3][N]) and to output tuple slot 2.
// ---------------------------------------------------------------------------
__global__ void __launch_bounds__(256)
warp_knn3_kernel(const float* __restrict__ xyz1,
                 const float* __restrict__ xyz2,
                 const float* __restrict__ flow1,
                 float* __restrict__ warped_ws,
                 float* __restrict__ warped_out)
{
    __shared__ float sx[256], sy[256], sz[256];
    const int b = blockIdx.y;
    const int j = blockIdx.x * 256 + threadIdx.x;
    const float* x1 = xyz1  + (size_t)b * 3 * NPTS;
    const float* f1 = flow1 + (size_t)b * 3 * NPTS;
    const float* x2 = xyz2  + (size_t)b * 3 * NPTS;

    const float qx = x2[0 * NPTS + j];
    const float qy = x2[1 * NPTS + j];
    const float qz = x2[2 * NPTS + j];

    float d0 = BIGF, d1 = BIGF, d2 = BIGF;
    int   i0 = 0,    i1 = 0,    i2 = 0;

    for (int tile = 0; tile < NPTS; tile += 256) {
        __syncthreads();
        const int c = tile + threadIdx.x;
        sx[threadIdx.x] = x1[0 * NPTS + c] + f1[0 * NPTS + c];
        sy[threadIdx.x] = x1[1 * NPTS + c] + f1[1 * NPTS + c];
        sz[threadIdx.x] = x1[2 * NPTS + c] + f1[2 * NPTS + c];
        if (tile + 256 < NPTS) {
            __builtin_prefetch(&x1[tile + 256 + threadIdx.x], 0, 0);
            __builtin_prefetch(&f1[tile + 256 + threadIdx.x], 0, 0);
        }
        __syncthreads();
        for (int t = 0; t < 256; ++t) {
            const float dx = sx[t] - qx;
            const float dy = sy[t] - qy;
            const float dz = sz[t] - qz;
            const float d  = dx * dx + dy * dy + dz * dz;
            const int  ci  = tile + t;
            if (d < d2) {
                if (d < d1) {
                    if (d < d0) { d2 = d1; i2 = i1; d1 = d0; i1 = i0; d0 = d; i0 = ci; }
                    else        { d2 = d1; i2 = i1; d1 = d;  i1 = ci; }
                } else          { d2 = d;  i2 = ci; }
            }
        }
    }

    int ids[3] = { i0, i1, i2 };
    float wk[3], wsum = 0.f;
    #pragma unroll
    for (int k = 0; k < 3; ++k) {
        const int i = ids[k];
        const float px = x1[0 * NPTS + i] + f1[0 * NPTS + i];
        const float py = x1[1 * NPTS + i] + f1[1 * NPTS + i];
        const float pz = x1[2 * NPTS + i] + f1[2 * NPTS + i];
        const float gx = px - qx, gy = py - qy, gz = pz - qz;
        float dist = sqrtf(gx * gx + gy * gy + gz * gz);
        dist = fmaxf(dist, 1e-10f);
        wk[k] = 1.0f / dist;
        wsum += wk[k];
    }
    float fx = 0.f, fy = 0.f, fz = 0.f;
    #pragma unroll
    for (int k = 0; k < 3; ++k) {
        const float w = wk[k] / wsum;
        const int i = ids[k];
        fx += w * f1[0 * NPTS + i];
        fy += w * f1[1 * NPTS + i];
        fz += w * f1[2 * NPTS + i];
    }
    const float wx = qx - fx, wy = qy - fy, wz = qz - fz;
    warped_ws [((size_t)b * 3 + 0) * NPTS + j] = wx;
    warped_ws [((size_t)b * 3 + 1) * NPTS + j] = wy;
    warped_ws [((size_t)b * 3 + 2) * NPTS + j] = wz;
    warped_out[((size_t)b * 3 + 0) * NPTS + j] = wx;
    warped_out[((size_t)b * 3 + 1) * NPTS + j] = wy;
    warped_out[((size_t)b * 3 + 2) * NPTS + j] = wz;
}

// ---------------------------------------------------------------------------
// Kernel B: 16-NN of each xyz1 point among warped_xyz2.  Unsorted register
// top-16 (downstream max-over-K is permutation invariant).
// ---------------------------------------------------------------------------
__global__ void __launch_bounds__(256)
knn16_kernel(const float* __restrict__ xyz1,
             const float* __restrict__ warped,
             int* __restrict__ idx_ws)
{
    __shared__ float sx[256], sy[256], sz[256];
    const int b = blockIdx.y;
    const int i = blockIdx.x * 256 + threadIdx.x;
    const float* x1 = xyz1   + (size_t)b * 3 * NPTS;
    const float* wp = warped + (size_t)b * 3 * NPTS;

    const float qx = x1[0 * NPTS + i];
    const float qy = x1[1 * NPTS + i];
    const float qz = x1[2 * NPTS + i];

    float bd[KNN]; int bi[KNN];
    #pragma unroll
    for (int u = 0; u < KNN; ++u) { bd[u] = BIGF; bi[u] = 0; }
    float wmax = BIGF;

    for (int tile = 0; tile < NPTS; tile += 256) {
        __syncthreads();
        const int c = tile + threadIdx.x;
        sx[threadIdx.x] = wp[0 * NPTS + c];
        sy[threadIdx.x] = wp[1 * NPTS + c];
        sz[threadIdx.x] = wp[2 * NPTS + c];
        if (tile + 256 < NPTS) {
            __builtin_prefetch(&wp[tile + 256 + threadIdx.x], 0, 0);
        }
        __syncthreads();
        for (int t = 0; t < 256; ++t) {
            const float dx = sx[t] - qx;
            const float dy = sy[t] - qy;
            const float dz = sz[t] - qz;
            const float d  = dx * dx + dy * dy + dz * dz;
            const int  ci  = tile + t;
            if (d < wmax) {
                bool done = false;
                #pragma unroll
                for (int u = 0; u < KNN; ++u) {
                    if (!done && bd[u] == wmax) { bd[u] = d; bi[u] = ci; done = true; }
                }
                wmax = bd[0];
                #pragma unroll
                for (int u = 1; u < KNN; ++u) wmax = fmaxf(wmax, bd[u]);
            }
        }
    }

    int* op = idx_ws + ((size_t)b * NPTS + i) * KNN;
    #pragma unroll
    for (int u = 0; u < KNN; ++u) op[u] = bi[u];
}

// ---------------------------------------------------------------------------
// Kernel C: pack weights to f16 B-fragment-friendly layouts.
//   w1h: [HID][KPAD]  (zero-padded contraction 67 -> 96)
//   w2h: [OUTC][HID]
// ---------------------------------------------------------------------------
__global__ void prep_weights_kernel(const float* __restrict__ W1,
                                    const float* __restrict__ W2,
                                    _Float16* __restrict__ w1h,
                                    _Float16* __restrict__ w2h)
{
    const int tid    = blockIdx.x * blockDim.x + threadIdx.x;
    const int stride = gridDim.x * blockDim.x;
    for (int e = tid; e < HID * KPAD; e += stride) {
        const int n = e / KPAD, c = e % KPAD;
        w1h[e] = (c < 3 + DFEAT) ? (_Float16)W1[n * (3 + DFEAT) + c] : (_Float16)0.f;
    }
    for (int e = tid; e < OUTC * HID; e += stride) {
        w2h[e] = (_Float16)W2[e];
    }
}

// ---------------------------------------------------------------------------
// Kernel D: one wave == one point.  Weights staged in LDS once per block
// (B-fragments become ds_load_b128).  Per-wave staging buffer is aliased
// between A1 (16x96 f16) and H (16x128 f16).  40 v_wmma per point, leaky
// between layers, max over the 16 neighbor rows, write [B,64,N].
// ---------------------------------------------------------------------------
__global__ void __launch_bounds__(WVB * 32)
mlp_kernel(const float* __restrict__ xyz1,
           const float* __restrict__ feat2,
           const float* __restrict__ warped,
           const int* __restrict__ idx_ws,
           const _Float16* __restrict__ w1h,
           const _Float16* __restrict__ w2h,
           const float* __restrict__ b1,
           const float* __restrict__ b2,
           float* __restrict__ out)
{
    __shared__ __attribute__((aligned(32))) _Float16 stage[WVB][KNN * HID]; // 16 KB (A1/H aliased)
    __shared__ __attribute__((aligned(32))) _Float16 w1s[HID * KPAD];       // 24 KB
    __shared__ __attribute__((aligned(32))) _Float16 w2s[OUTC * HID];       // 16 KB
    __shared__ int idxs[WVB][KNN];

    // ---- cooperative weight staging: global f16 -> LDS (uint4 copies) ----
    {
        const uint4* s1 = (const uint4*)w1h;
        const uint4* s2 = (const uint4*)w2h;
        uint4* d1 = (uint4*)w1s;
        uint4* d2 = (uint4*)w2s;
        const int n1 = HID * KPAD / 8;   // 1536 uint4
        const int n2 = OUTC * HID / 8;   // 1024 uint4
        for (int i = threadIdx.x; i < n1; i += WVB * 32) d1[i] = s1[i];
        for (int i = threadIdx.x; i < n2; i += WVB * 32) d2[i] = s2[i];
    }
    __syncthreads();

    const int wv   = threadIdx.x >> 5;
    const int lane = threadIdx.x & 31;
    const int pt   = blockIdx.x * WVB + wv;
    const int b    = pt / NPTS;
    const int n    = pt % NPTS;

    const float* x1 = xyz1 + (size_t)b * 3 * NPTS;
    const float q[3] = { x1[0 * NPTS + n], x1[1 * NPTS + n], x1[2 * NPTS + n] };

    if (lane < KNN) idxs[wv][lane] = idx_ws[((size_t)b * NPTS + n) * KNN + lane];
    asm volatile("s_wait_dscnt 0x0" ::: "memory");

    // ---- gather A1 = [xyz_norm | feat] into stage, f16, padded to KPAD ----
    const float* ft = feat2  + (size_t)b * DFEAT * NPTS;
    const float* wp = warped + (size_t)b * 3 * NPTS;
    for (int e = lane; e < KNN * KPAD; e += 32) {
        const int m = e / KPAD, c = e % KPAD;
        const int j = idxs[wv][m];
        float v;
        if (c < 3)              v = wp[c * NPTS + j] - q[c];
        else if (c < 3 + DFEAT) v = ft[(c - 3) * NPTS + j];
        else                    v = 0.f;
        stage[wv][m * KPAD + c] = (_Float16)v;
    }
    asm volatile("s_wait_dscnt 0x0" ::: "memory");

    const int row = lane & 15;
    const int hi  = lane >> 4;   // 0: K low half, 1: K high half (A layout)

    // ---- load layer-1 A fragments (16x32 f16 each) into registers ----
    v16h afrag[3];
    #pragma unroll
    for (int kt = 0; kt < 3; ++kt) {
        const int kofs = kt * 32 + hi * 16;
        v8h lo = *(const v8h*)&stage[wv][row * KPAD + kofs];
        v8h hh = *(const v8h*)&stage[wv][row * KPAD + kofs + 8];
        #pragma unroll
        for (int u = 0; u < 8; ++u) { afrag[kt][u] = lo[u]; afrag[kt][u + 8] = hh[u]; }
    }
    // A1 fully consumed into registers; stage[] may now be reused for H.
    asm volatile("s_wait_dscnt 0x0" ::: "memory");

    // ---- layer 1: [16x96] x [96x128] -> leaky -> H (f16) in stage ----
    #pragma unroll
    for (int t = 0; t < 8; ++t) {
        const float bias = b1[t * 16 + row];
        v8f acc;
        #pragma unroll
        for (int r = 0; r < 8; ++r) acc[r] = bias;
        #pragma unroll
        for (int kt = 0; kt < 3; ++kt) {
            const v16h bfrag = *(const v16h*)&w1s[(t * 16 + row) * KPAD + kt * 32 + hi * 16];
            acc = __builtin_amdgcn_wmma_f32_16x16x32_f16(
                false, afrag[kt], false, bfrag, (short)0, acc, false, false);
        }
        #pragma unroll
        for (int r = 0; r < 8; ++r) {
            float v = acc[r];
            v = (v >= 0.f) ? v : 0.1f * v;
            stage[wv][(r + hi * 8) * HID + t * 16 + row] = (_Float16)v;
        }
    }
    asm volatile("s_wait_dscnt 0x0" ::: "memory");

    // ---- load layer-2 A fragments from H ----
    v16h a2[4];
    #pragma unroll
    for (int kt = 0; kt < 4; ++kt) {
        const int kofs = kt * 32 + hi * 16;
        v8h lo = *(const v8h*)&stage[wv][row * HID + kofs];
        v8h hh = *(const v8h*)&stage[wv][row * HID + kofs + 8];
        #pragma unroll
        for (int u = 0; u < 8; ++u) { a2[kt][u] = lo[u]; a2[kt][u + 8] = hh[u]; }
    }

    // ---- layer 2: [16x128] x [128x64] -> leaky -> max over 16 rows ----
    #pragma unroll
    for (int t = 0; t < 4; ++t) {
        const float bias = b2[t * 16 + row];
        v8f acc;
        #pragma unroll
        for (int r = 0; r < 8; ++r) acc[r] = bias;
        #pragma unroll
        for (int kt = 0; kt < 4; ++kt) {
            const v16h bfrag = *(const v16h*)&w2s[(t * 16 + row) * HID + kt * 32 + hi * 16];
            acc = __builtin_amdgcn_wmma_f32_16x16x32_f16(
                false, a2[kt], false, bfrag, (short)0, acc, false, false);
        }
        float mx = -BIGF;
        #pragma unroll
        for (int r = 0; r < 8; ++r) {
            float v = acc[r];
            v = (v >= 0.f) ? v : 0.1f * v;
            mx = fmaxf(mx, v);
        }
        const float other = __shfl_xor(mx, 16, 32);   // combine M halves
        mx = fmaxf(mx, other);
        if (hi == 0) {
            out[(size_t)(b * OUTC + t * 16 + row) * NPTS + n] = mx;
        }
    }
}

// ---------------------------------------------------------------------------
extern "C" void kernel_launch(void* const* d_in, const int* in_sizes, int n_in,
                              void* d_out, int out_size, void* d_ws, size_t ws_size,
                              hipStream_t stream) {
    const float* xyz1  = (const float*)d_in[0];
    const float* xyz2  = (const float*)d_in[1];
    const float* flow1 = (const float*)d_in[2];
    const float* feat2 = (const float*)d_in[3];
    const float* W1    = (const float*)d_in[4];
    const float* b1    = (const float*)d_in[5];
    const float* W2    = (const float*)d_in[6];
    const float* b2    = (const float*)d_in[7];
    // d_in[8] = nsample (==16, compiled in as KNN)

    char* ws = (char*)d_ws;
    const size_t warped_bytes = (size_t)BATCH * 3 * NPTS * sizeof(float);     // 196608
    const size_t idx_bytes    = (size_t)BATCH * NPTS * KNN * sizeof(int);     // 1048576
    float*    warped = (float*)ws;
    int*      idx16  = (int*)(ws + warped_bytes);
    _Float16* w1h    = (_Float16*)(ws + warped_bytes + idx_bytes);
    _Float16* w2h    = (_Float16*)(ws + warped_bytes + idx_bytes
                                   + (size_t)HID * KPAD * sizeof(_Float16));

    float* out        = (float*)d_out;                       // [B,64,N]
    float* warped_out = out + (size_t)BATCH * OUTC * NPTS;   // [B,3,N]

    dim3 gKNN(NPTS / 256, BATCH);
    warp_knn3_kernel<<<gKNN, 256, 0, stream>>>(xyz1, xyz2, flow1, warped, warped_out);
    knn16_kernel   <<<gKNN, 256, 0, stream>>>(xyz1, warped, idx16);
    prep_weights_kernel<<<48, 256, 0, stream>>>(W1, W2, w1h, w2h);
    mlp_kernel<<<(BATCH * NPTS) / WVB, WVB * 32, 0, stream>>>(
        xyz1, feat2, warped, idx16, w1h, w2h, b1, b2, out);
}
